// Attention_87574383166081
// MI455X (gfx1250) — compile-verified, matrix operand order checked by reading, not run on hardware
//
#include <hip/hip_runtime.h>
#include <math.h>

// ---------------- types & helpers ----------------
typedef __attribute__((ext_vector_type(16))) __bf16          v16bf;
typedef __attribute__((ext_vector_type(16))) unsigned short  v16us;
typedef __attribute__((ext_vector_type(8)))  float           v8f;

union BFrag { v16us u; v16bf b; };

__device__ __forceinline__ unsigned short f32_to_bf16(float f) {
  unsigned int x = __float_as_uint(f);
  unsigned int r = x + 0x7FFFu + ((x >> 16) & 1u);   // round-to-nearest-even
  return (unsigned short)(r >> 16);
}

// Native packed f32->bf16 conversion (single VOP3): v_cvt_pk_bf16_f32.
// Clang does not declare the builtin for gfx1250, but the instruction is part
// of the CDNA4+ bf16-cvt family; emit it directly.
__device__ __forceinline__ unsigned int pack_bf16x2(float lo, float hi) {
  unsigned int r;
  asm("v_cvt_pk_bf16_f32 %0, %1, %2" : "=v"(r) : "v"(lo), "v"(hi));
  return r;
}

// K-offset of element-pair e (e=0..7) for 16-bit WMMA A/B fragments (ISA 7.12.2)
__device__ __forceinline__ int kkof(int e, int hw) {
  return ((e & 3) << 1) + ((e >> 2) << 4) + (hw << 3);
}

// gfx1250 async copy global->LDS, 16 bytes per lane; tracked by ASYNCcnt.
__device__ __forceinline__ void async_g2l_b128(unsigned int lds_off,
                                               const void* gaddr) {
  asm volatile("global_load_async_to_lds_b128 %0, %1, off"
               :: "v"(lds_off), "v"((unsigned long long)(uintptr_t)gaddr)
               : "memory");
}
__device__ __forceinline__ void wait_asynccnt0() {
  asm volatile("s_wait_asynccnt 0x0" ::: "memory");
}

// ---------------- GEMM: Out[M][N] = A[M][K] @ W[N][K]^T  (bf16 WMMA, f32 acc) ----------------
#define BM 128
#define BN 128
#define BK 32
#define LDT 36   // padded LDS row stride (ushort units)

__global__ __launch_bounds__(256) void gemm_bf16_wmma(
    const float* __restrict__ A, int K,
    const float* __restrict__ W0, int N0,
    const float* __restrict__ W1, int N1,
    const float* __restrict__ W2,
    float* __restrict__ Out, int ldo)
{
  __shared__ unsigned short As[BM * LDT];
  __shared__ unsigned short Ws[BN * LDT];

  const int tid  = threadIdx.x;
  const int lane = tid & 31;
  const int wave = tid >> 5;          // 0..7  (2 N-waves x 4 M-waves)
  const int l16  = lane & 15;
  const int hw   = lane >> 4;

  const int m0  = blockIdx.y * BM;
  const int n0g = blockIdx.x * BN;

  // pick weight segment (column tiles never straddle segments: all multiples of 128)
  const float* W; int nw;
  if (n0g < N0)           { W = W0; nw = n0g; }
  else if (n0g < N0 + N1) { W = W1; nw = n0g - N0; }
  else                    { W = W2; nw = n0g - N0 - N1; }

  const int wm = (wave >> 1) * 32;    // wave M offset: 0,32,64,96
  const int wn = (wave & 1) * 64;     // wave N offset: 0,64

  v8f acc[2][4];
  v8f zero = {0.f,0.f,0.f,0.f,0.f,0.f,0.f,0.f};
#pragma unroll
  for (int i = 0; i < 2; ++i)
#pragma unroll
    for (int j = 0; j < 4; ++j) acc[i][j] = zero;

  const int srow = tid >> 1;          // staging row 0..127
  const int skb  = (tid & 1) * 16;    // staging k-half

  const float* arow = A + (size_t)(m0 + srow) * K + skb;
  const float* wrow = W + (size_t)(nw + srow) * K + skb;

  // register prefetch of first k-tile (software pipeline over HBM latency)
  float4 ra[4], rw[4];
  {
    const float4* ap = (const float4*)arow;
    const float4* wp = (const float4*)wrow;
#pragma unroll
    for (int q = 0; q < 4; ++q) { ra[q] = ap[q]; rw[q] = wp[q]; }
  }

  for (int kt = 0; kt < K; kt += BK) {
    __syncthreads();
    {
      unsigned int* du = (unsigned int*)&As[srow * LDT + skb];
      du[0] = pack_bf16x2(ra[0].x, ra[0].y); du[1] = pack_bf16x2(ra[0].z, ra[0].w);
      du[2] = pack_bf16x2(ra[1].x, ra[1].y); du[3] = pack_bf16x2(ra[1].z, ra[1].w);
      du[4] = pack_bf16x2(ra[2].x, ra[2].y); du[5] = pack_bf16x2(ra[2].z, ra[2].w);
      du[6] = pack_bf16x2(ra[3].x, ra[3].y); du[7] = pack_bf16x2(ra[3].z, ra[3].w);
      unsigned int* dw = (unsigned int*)&Ws[srow * LDT + skb];
      dw[0] = pack_bf16x2(rw[0].x, rw[0].y); dw[1] = pack_bf16x2(rw[0].z, rw[0].w);
      dw[2] = pack_bf16x2(rw[1].x, rw[1].y); dw[3] = pack_bf16x2(rw[1].z, rw[1].w);
      dw[4] = pack_bf16x2(rw[2].x, rw[2].y); dw[5] = pack_bf16x2(rw[2].z, rw[2].w);
      dw[6] = pack_bf16x2(rw[3].x, rw[3].y); dw[7] = pack_bf16x2(rw[3].z, rw[3].w);
    }
    __syncthreads();

    // issue next tile's global loads; they retire while this tile's WMMAs run
    if (kt + BK < K) {
      const float4* ap = (const float4*)(arow + kt + BK);
      const float4* wp = (const float4*)(wrow + kt + BK);
#pragma unroll
      for (int q = 0; q < 4; ++q) { ra[q] = ap[q]; rw[q] = wp[q]; }
    }

    BFrag afr[2], bfr[4];
#pragma unroll
    for (int i = 0; i < 2; ++i) {
      const int row = wm + i * 16 + l16;
      unsigned int* dst = (unsigned int*)&afr[i].u;
#pragma unroll
      for (int e = 0; e < 8; ++e)
        dst[e] = *(const unsigned int*)&As[row * LDT + kkof(e, hw)];
    }
#pragma unroll
    for (int j = 0; j < 4; ++j) {
      const int col = wn + j * 16 + l16;
      unsigned int* dst = (unsigned int*)&bfr[j].u;
#pragma unroll
      for (int e = 0; e < 8; ++e)
        dst[e] = *(const unsigned int*)&Ws[col * LDT + kkof(e, hw)];
    }
#pragma unroll
    for (int i = 0; i < 2; ++i)
#pragma unroll
      for (int j = 0; j < 4; ++j)
        acc[i][j] = __builtin_amdgcn_wmma_f32_16x16x32_bf16(
            false, afr[i].b, false, bfr[j].b, (short)0, acc[i][j], false, false);
  }

#pragma unroll
  for (int i = 0; i < 2; ++i)
#pragma unroll
    for (int j = 0; j < 4; ++j)
#pragma unroll
      for (int r = 0; r < 8; ++r) {
        const int m = m0 + wm + i * 16 + 8 * hw + r;
        const int n = n0g + wn + j * 16 + l16;
        Out[(size_t)m * ldo + n] = acc[i][j][r];
      }
}

// ---------------- RMSNorm + RoPE + bf16 pack ----------------
// one wave per (s, "head"):  h<32: Q head, 32..39: K head, 40..47: V head (pack only, transposed)
__global__ __launch_bounds__(256) void rmsnorm_rope_pack(
    const float* __restrict__ qkv,
    const float* __restrict__ qw, const float* __restrict__ kw,
    const float* __restrict__ cosT, const float* __restrict__ sinT,
    const int* __restrict__ positions,
    unsigned short* __restrict__ Qb,   // [32][2048][128]
    unsigned short* __restrict__ Kb,   // [8][2048][128]
    unsigned short* __restrict__ Vt)   // [8][128][2048]  (transposed)
{
  const int gw   = blockIdx.x * 8 + (threadIdx.x >> 5);
  const int lane = threadIdx.x & 31;
  const int s = gw / 48;
  const int h = gw % 48;
  const int dbase = lane * 4;

  if (h < 40) {
    const int col = (h < 32) ? h * 128 : 4096 + (h - 32) * 128;
    const float4 xv = *(const float4*)(qkv + (size_t)s * 6144 + col + dbase);
    float x[4] = {xv.x, xv.y, xv.z, xv.w};
    float ss = x[0]*x[0] + x[1]*x[1] + x[2]*x[2] + x[3]*x[3];
#pragma unroll
    for (int m = 16; m >= 1; m >>= 1) ss += __shfl_xor(ss, m, 32);
    const float inv = rsqrtf(ss * (1.0f / 128.0f) + 1e-6f);
    const float* nwp = (h < 32) ? qw : kw;
    const int pos = positions[s];
    float xn[4], xp[4], ro[4];
#pragma unroll
    for (int c = 0; c < 4; ++c) xn[c] = x[c] * inv * nwp[dbase + c];
#pragma unroll
    for (int c = 0; c < 4; ++c) xp[c] = __shfl_xor(xn[c], 16, 32);  // partner holds d +/- 64
#pragma unroll
    for (int c = 0; c < 4; ++c) {
      const int d = dbase + c;
      const float cs = cosT[pos * 128 + d];
      const float sn = sinT[pos * 128 + d];
      const float rot = (d < 64) ? -xp[c] : xp[c];
      ro[c] = xn[c] * cs + rot * sn;
    }
    unsigned short* dst = (h < 32)
        ? Qb + ((size_t)h * 2048 + s) * 128 + dbase
        : Kb + ((size_t)(h - 32) * 2048 + s) * 128 + dbase;
    *(uint2*)dst = make_uint2(pack_bf16x2(ro[0], ro[1]), pack_bf16x2(ro[2], ro[3]));
  } else {
    const int kv = h - 40;
    const float4 xv = *(const float4*)(qkv + (size_t)s * 6144 + 5120 + kv * 128 + dbase);
    const float x[4] = {xv.x, xv.y, xv.z, xv.w};
#pragma unroll
    for (int c = 0; c < 4; ++c)
      Vt[((size_t)kv * 128 + dbase + c) * 2048 + s] = f32_to_bf16(x[c]);
  }
}

// ---------------- causal flash attention (bf16 WMMA, online softmax) ----------------
// LDS is DYNAMIC so the raw byte offsets passed to global_load_async_to_lds are
// real offsets from the wave's LDS base (no static LDS in this kernel).
#define AKP 136   // Ks   LDS stride (ushorts): 32 keys x 128 d   (row = 272 B, 16B-aligned)
#define AVP 40    // Vts  LDS stride (ushorts): 128 d  x 32 keys  (row =  80 B, 16B-aligned)
#define KS_BYTES   (32 * AKP * 2)              // 8704
#define ATTN_LDS   (KS_BYTES + 128 * AVP * 2)  // 18944

__global__ __launch_bounds__(128) void flash_attn_wmma(
    const unsigned short* __restrict__ Qb,
    const unsigned short* __restrict__ Kb,
    const unsigned short* __restrict__ Vt,
    float* __restrict__ attnout)     // [2048][4096] = [s][h*128+d]
{
  extern __shared__ unsigned short smem_us[];
  unsigned short* Ks  = smem_us;                 // LDS byte offset 0
  unsigned short* Vts = smem_us + 32 * AKP;      // LDS byte offset KS_BYTES

  const int h  = blockIdx.x;        // 0..31
  const int qb = blockIdx.y;        // 0..31 (64 queries each)
  const int kv = h >> 2;            // GQA group of 4
  const int tid = threadIdx.x;
  const int lane = tid & 31, wave = tid >> 5;
  const int l16 = lane & 15, hw = lane >> 4;

  const int q0   = qb * 64 + wave * 16;
  const int qcol = q0 + l16;        // this lane's global query index

  // Q^T B-fragments (4 chunks of 32 along d), held in registers for whole loop
  BFrag qf[4];
  {
    const unsigned short* qrow = Qb + ((size_t)h * 2048 + qcol) * 128;
#pragma unroll
    for (int c = 0; c < 4; ++c) {
      unsigned int* dst = (unsigned int*)&qf[c].u;
#pragma unroll
      for (int e = 0; e < 8; ++e)
        dst[e] = *(const unsigned int*)(qrow + 32 * c + kkof(e, hw));
    }
  }

  v8f o[8];
  v8f zero = {0.f,0.f,0.f,0.f,0.f,0.f,0.f,0.f};
#pragma unroll
  for (int t = 0; t < 8; ++t) o[t] = zero;
  float mrow = -INFINITY, lrow = 0.f;
  const float scale = 0.08838834764831845f;   // 1/sqrt(128)

  // per-thread staging coordinates
  const int krow = tid >> 2, kseg = (tid & 3) * 32;   // K: 32 rows x 128 elems
  const unsigned short* kbase = Kb + ((size_t)kv * 2048 + krow) * 128 + kseg;
  const unsigned short* vbase = Vt + ((size_t)kv * 128 + tid) * 2048;
  const unsigned int klds = (unsigned int)(krow * AKP + kseg) * 2;
  const unsigned int vlds = (unsigned int)KS_BYTES + (unsigned int)tid * (AVP * 2);

  const int nkb = qb * 2 + 2;                 // key blocks up to diagonal
  for (int kbi = 0; kbi < nkb; ++kbi) {
    const int kb0 = kbi * 32;
    __syncthreads();                          // LDS consumers of prev block done
    // async-copy K block (32x128 bf16) and V^T block (128x32 bf16) into LDS:
    // 8 x b128 per thread, no VGPR round trip, tracked by ASYNCcnt.
#pragma unroll
    for (int j = 0; j < 4; ++j)
      async_g2l_b128(klds + j * 16, kbase + (size_t)kb0 * 128 + j * 8);
#pragma unroll
    for (int j = 0; j < 4; ++j)
      async_g2l_b128(vlds + j * 16, vbase + kb0 + j * 8);
    wait_asynccnt0();
    __syncthreads();                          // all waves' copies visible

    // S^T = K(32x128) @ Q^T(128x16): two 16x16 C tiles (keys as M, queries as N)
    v8f st[2]; st[0] = zero; st[1] = zero;
#pragma unroll
    for (int c = 0; c < 4; ++c)
#pragma unroll
      for (int j = 0; j < 2; ++j) {
        BFrag ak;
        unsigned int* dst = (unsigned int*)&ak.u;
        const int kr = j * 16 + l16;
#pragma unroll
        for (int e = 0; e < 8; ++e)
          dst[e] = *(const unsigned int*)&Ks[kr * AKP + 32 * c + kkof(e, hw)];
        st[j] = __builtin_amdgcn_wmma_f32_16x16x32_bf16(
            false, ak.b, false, qf[c].b, (short)0, st[j], false, false);
      }

    // scale + causal mask + online softmax
    float vals[16];
    float pm = -INFINITY;
#pragma unroll
    for (int j = 0; j < 2; ++j)
#pragma unroll
      for (int r = 0; r < 8; ++r) {
        const int kg = kb0 + j * 16 + 8 * hw + r;
        const float sv = st[j][r] * scale;
        const float v = (kg <= qcol) ? sv : -INFINITY;
        vals[j * 8 + r] = v;
        pm = fmaxf(pm, v);
      }
    pm = fmaxf(pm, __shfl_xor(pm, 16, 32));
    const float mnew = fmaxf(mrow, pm);
    const float f = __expf(mrow - mnew);      // first iter: exp(-inf)=0
    lrow *= f;
#pragma unroll
    for (int t = 0; t < 8; ++t)
#pragma unroll
      for (int r = 0; r < 8; ++r) o[t][r] *= f;

    // P in bf16 B-fragment layout: C-layout of (st[0],st[1]) maps 1:1 onto a
    // 32x16 bf16 B operand; exp + pairwise pack with v_cvt_pk_bf16_f32.
    BFrag pf;
    float psum = 0.f;
    {
      unsigned int* pu = (unsigned int*)&pf.u;
#pragma unroll
      for (int i = 0; i < 8; ++i) {
        const float p0 = __expf(vals[2 * i]     - mnew);  // masked -> 0
        const float p1 = __expf(vals[2 * i + 1] - mnew);
        psum += p0 + p1;
        pu[i] = pack_bf16x2(p0, p1);
      }
    }
    psum += __shfl_xor(psum, 16, 32);
    lrow += psum;
    mrow = mnew;

    // O^T(128x16) += V^T(128x32) @ P^T(32x16)
#pragma unroll
    for (int t = 0; t < 8; ++t) {
      BFrag av;
      unsigned int* dst = (unsigned int*)&av.u;
      const int drow = t * 16 + l16;
#pragma unroll
      for (int e = 0; e < 8; ++e)
        dst[e] = *(const unsigned int*)&Vts[drow * AVP + kkof(e, hw)];
      o[t] = __builtin_amdgcn_wmma_f32_16x16x32_bf16(
          false, av.b, false, pf.b, (short)0, o[t], false, false);
    }
  }

  const float inv = 1.0f / lrow;
#pragma unroll
  for (int t = 0; t < 8; ++t)
#pragma unroll
    for (int r = 0; r < 8; ++r) {
      const int d = t * 16 + 8 * hw + r;
      attnout[(size_t)qcol * 4096 + h * 128 + d] = o[t][r] * inv;
    }
}

// ---------------- launcher ----------------
extern "C" void kernel_launch(void* const* d_in, const int* in_sizes, int n_in,
                              void* d_out, int out_size, void* d_ws, size_t ws_size,
                              hipStream_t stream) {
  const float* x    = (const float*)d_in[0];
  const float* wq   = (const float*)d_in[1];
  const float* wk   = (const float*)d_in[2];
  const float* wv   = (const float*)d_in[3];
  const float* wo   = (const float*)d_in[4];
  const float* qnw  = (const float*)d_in[5];
  const float* knw  = (const float*)d_in[6];
  const float* cosT = (const float*)d_in[7];
  const float* sinT = (const float*)d_in[8];
  const int*   pos  = (const int*)d_in[11];
  float* out = (float*)d_out;

  char* ws = (char*)d_ws;
  float*          qkv = (float*)ws;                                  // 2048*6144 f32  = 48 MB
  unsigned short* Qb  = (unsigned short*)(ws + 50331648);            // 32*2048*128 bf16 = 16 MB
  unsigned short* Kb  = (unsigned short*)(ws + 50331648 + 16777216); // 8*2048*128 bf16  = 4 MB
  unsigned short* Vt  = (unsigned short*)(ws + 50331648 + 16777216 + 4194304);
  float*      attnout = (float*)(ws + 50331648 + 16777216 + 4194304 + 4194304);

  // 1) fused QKV projection: [2048][6144] = x @ [wq;wk;wv]^T
  gemm_bf16_wmma<<<dim3(48, 16), 256, 0, stream>>>(x, 4096, wq, 4096, wk, 1024, wv,
                                                   qkv, 6144);
  // 2) per-head RMSNorm + RoPE + bf16 pack (V transposed)
  rmsnorm_rope_pack<<<12288, 256, 0, stream>>>(qkv, qnw, knw, cosT, sinT, pos,
                                               Qb, Kb, Vt);
  // 3) causal flash attention (dynamic LDS for async-to-LDS copies)
  flash_attn_wmma<<<dim3(32, 32), 128, ATTN_LDS, stream>>>(Qb, Kb, Vt, attnout);
  // 4) output projection: out = attnout @ wo^T
  gemm_bf16_wmma<<<dim3(32, 16), 256, 0, stream>>>(attnout, 4096, wo, 4096, wo, 0, wo,
                                                   out, 4096);
}
